// TargetAttention_89928025244344
// MI455X (gfx1250) — matrix operand order repaired and checked by reading, not compiled, not run on hardware
//
#include <hip/hip_runtime.h>

// ---------------------------------------------------------------------------
// Types for CDNA5 WMMA (wave32, 16x16x32 bf16 -> f32)
// ---------------------------------------------------------------------------
typedef __bf16          v16bf __attribute__((ext_vector_type(16)));
typedef unsigned short  v16us __attribute__((ext_vector_type(16)));
typedef float           v8f   __attribute__((ext_vector_type(8)));

__device__ __forceinline__ unsigned short f2bf(float f) {
    // round-to-nearest-even f32 -> bf16
    unsigned int u = __float_as_uint(f);
    unsigned int r = 0x7FFFu + ((u >> 16) & 1u);
    u += r;
    return (unsigned short)(u >> 16);
}

// Problem constants
#define BATCH 2048
#define SEQ   200
#define NTILE 13       // ceil(200/16) S-tiles
#define DIM   128
#define H1    256
#define H2    128
#define FEAT  512      // 4*DIM

#define KSTRIDE  132   // padded key-row stride (floats): conflict-free LDS
#define H1STRIDE 264   // padded h1 row stride (ushorts)

// ---------------------------------------------------------------------------
// Prep: convert W1 (512x256) and W2 (256x128) f32 -> bf16, pre-swizzled into
// the exact WMMA B-fragment layout:
//   frag(nt,kc): lane L holds column n = nt*16 + (L&15),
//   K values k = kc*32 + (L<16 ? 0 : 8) + (i<8 ? i : i+8),  i = 0..15
// ---------------------------------------------------------------------------
__global__ __launch_bounds__(256) void prep_swizzle(
    const float* __restrict__ W1, const float* __restrict__ W2,
    unsigned short* __restrict__ w1sw, unsigned short* __restrict__ w2sw)
{
    int idx = blockIdx.x * 256 + threadIdx.x;
    if (idx < FEAT * H1) {
        int i    = idx & 15;
        int lane = (idx >> 4) & 31;
        int frag = idx >> 9;
        int kc   = frag & 15;          // 16 K-chunks of 32 over FEAT=512
        int nt   = frag >> 4;          // 16 N-tiles over H1=256
        int n    = nt * 16 + (lane & 15);
        int k    = kc * 32 + ((lane < 16) ? 0 : 8) + (i < 8 ? i : i + 8);
        w1sw[idx] = f2bf(W1[k * H1 + n]);
    } else {
        int j = idx - FEAT * H1;
        if (j < H1 * H2) {
            int i    = j & 15;
            int lane = (j >> 4) & 31;
            int frag = j >> 9;
            int kc   = frag & 7;       // 8 K-chunks over H1=256
            int nt   = frag >> 3;      // 8 N-tiles over H2=128
            int n    = nt * 16 + (lane & 15);
            int k    = kc * 32 + ((lane < 16) ? 0 : 8) + (i < 8 ? i : i + 8);
            w2sw[j] = f2bf(W2[k * H2 + n]);
        }
    }
}

// ---------------------------------------------------------------------------
// Score kernel: ONE wave32 per (batch row, 16-row S-tile).
// 2048*13 = 26624 single-wave workgroups; ~17KB static LDS each, so many
// blocks co-reside per WGP and hide the B-fragment L2 latency.
// ---------------------------------------------------------------------------
__global__ __launch_bounds__(32) void attn_scores(
    const float* __restrict__ qg, const float* __restrict__ hk,
    const float* __restrict__ b1, const float* __restrict__ b2,
    const float* __restrict__ W3, const float* __restrict__ b3,
    const unsigned short* __restrict__ w1sw,
    const unsigned short* __restrict__ w2sw,
    float* __restrict__ scoresbuf)
{
    __shared__ float          kt[16 * KSTRIDE];     // 8448 B : this tile's keys
    __shared__ float          qb[DIM];              //  512 B : query
    __shared__ unsigned short h1[16 * H1STRIDE];    // 8448 B : relu(layer1) bf16

    const int blk  = blockIdx.x;
    const int b    = blk / NTILE;
    const int st   = blk % NTILE;
    const int lane = threadIdx.x;

    // ---- stage this tile's 16 key rows (zero-padded) + query into LDS ----
    const float4* hk4 = (const float4*)(hk + (size_t)b * SEQ * DIM);
    for (int idx = lane; idx < 16 * (DIM / 4); idx += 32) {
        int r  = idx >> 5;              // DIM/4 == 32
        int d4 = idx & 31;
        int s  = st * 16 + r;
        float4 v = make_float4(0.f, 0.f, 0.f, 0.f);
        if (s < SEQ) v = hk4[s * (DIM / 4) + d4];
        *(float4*)(kt + r * KSTRIDE + d4 * 4) = v;
    }
    for (int d = lane; d < DIM; d += 32) qb[d] = qg[(size_t)b * DIM + d];
    __threadfence_block();              // order LDS stores before reads (1 wave)

    const int mrow     = lane & 15;     // M row held by this lane (A row; C column)
    const int halfsel  = lane >> 4;     // 0: K 0-7/16-23, 1: K 8-15/24-31
    const int koffBase = halfsel ? 8 : 0;
    const float* krow  = kt + mrow * KSTRIDE;

    const v16us* w1v = (const v16us*)w1sw;
    const v16us* w2v = (const v16us*)w2sw;

    // ---- build layer-1 A fragments (16 x 512 bf16) in register layout ----
    v16bf a1[16];
    for (int kc = 0; kc < 16; ++kc) {
        const int sec   = kc >> 2;      // feature section, uniform per kc
        const int dbase = (kc & 3) * 32;
        v16us t;
#pragma unroll
        for (int i = 0; i < 16; i += 2) {
            int koff = koffBase + (i < 8 ? i : i + 8);
            int d    = dbase + koff;
            float q0 = qb[d],   q1 = qb[d + 1];
            float k0 = krow[d], k1 = krow[d + 1];
            float f0, f1;
            if      (sec == 0) { f0 = q0;      f1 = q1;      }
            else if (sec == 1) { f0 = k0;      f1 = k1;      }
            else if (sec == 2) { f0 = q0 - k0; f1 = q1 - k1; }
            else               { f0 = q0 * k0; f1 = q1 * k1; }
            t[i]     = f2bf(f0);
            t[i + 1] = f2bf(f1);
        }
        a1[kc] = __builtin_bit_cast(v16bf, t);
    }

    // ---- layer 1: h1 = relu(F @ W1 + b1) -> LDS (bf16) ----
    for (int nt = 0; nt < 16; ++nt) {
        v8f acc = {};
#pragma unroll
        for (int kc = 0; kc < 16; ++kc) {
            v16bf bb = __builtin_bit_cast(v16bf, w1v[(nt * 16 + kc) * 32 + lane]);
            acc = __builtin_amdgcn_wmma_f32_16x16x32_bf16(
                false, a1[kc], false, bb, (short)0, acc, false, false);
        }
        int   n    = nt * 16 + mrow;
        float bias = b1[n];
#pragma unroll
        for (int r = 0; r < 8; ++r) {
            float v = acc[r] + bias;
            v = v > 0.f ? v : 0.f;
            h1[(halfsel * 8 + r) * H1STRIDE + n] = f2bf(v);
        }
    }
    __threadfence_block();

    // ---- build layer-2 A fragments from h1 (16 x 256 bf16) ----
    v16bf a2[8];
    for (int kc = 0; kc < 8; ++kc) {
        v16us t;
#pragma unroll
        for (int i = 0; i < 16; i += 2) {
            int koff = koffBase + (i < 8 ? i : i + 8);
            int col  = kc * 32 + koff;
            unsigned int pk = *(const unsigned int*)(h1 + mrow * H1STRIDE + col);
            t[i]     = (unsigned short)(pk & 0xFFFFu);
            t[i + 1] = (unsigned short)(pk >> 16);
        }
        a2[kc] = __builtin_bit_cast(v16bf, t);
    }

    // ---- layer 2 + fused layer 3: sacc[r] = sum_n relu(h2+b2)*W3 ----
    float sacc[8] = {0.f, 0.f, 0.f, 0.f, 0.f, 0.f, 0.f, 0.f};
    for (int nt = 0; nt < 8; ++nt) {
        v8f acc = {};
#pragma unroll
        for (int kc = 0; kc < 8; ++kc) {
            v16bf bb = __builtin_bit_cast(v16bf, w2v[(nt * 8 + kc) * 32 + lane]);
            acc = __builtin_amdgcn_wmma_f32_16x16x32_bf16(
                false, a2[kc], false, bb, (short)0, acc, false, false);
        }
        int   n     = nt * 16 + mrow;
        float bias2 = b2[n];
        float w3v   = W3[n];
#pragma unroll
        for (int r = 0; r < 8; ++r) {
            float v = acc[r] + bias2;
            v = v > 0.f ? v : 0.f;
            sacc[r] += v * w3v;
        }
    }

    // ---- reduce each row across its 16-lane half; emit raw scores ----
    float bb3 = b3[0];
#pragma unroll
    for (int r = 0; r < 8; ++r) {
        float v = sacc[r];
        v += __shfl_xor(v, 1, 32);
        v += __shfl_xor(v, 2, 32);
        v += __shfl_xor(v, 4, 32);
        v += __shfl_xor(v, 8, 32);
        if (mrow == 0) {
            int s = st * 16 + halfsel * 8 + r;
            if (s < SEQ) scoresbuf[(size_t)b * SEQ + s] = v + bb3;
        }
    }
}

// ---------------------------------------------------------------------------
// Softmax + weighted key sum: one 256-thread block per batch row.
// ---------------------------------------------------------------------------
__global__ __launch_bounds__(256) void attn_softmax(
    const float* __restrict__ hk, const int* __restrict__ maskp,
    const float* __restrict__ scoresbuf, float* __restrict__ out)
{
    __shared__ float sc[256];
    __shared__ float red[256];

    const int b   = blockIdx.x;
    const int tid = threadIdx.x;

    // masked scores
    float v = -3.0e38f;
    if (tid < SEQ) {
        float s0 = scoresbuf[(size_t)b * SEQ + tid];
        int   mv = maskp[(size_t)b * SEQ + tid];
        v = mv ? s0 : -1e9f;
    }
    sc[tid] = v;
    __syncthreads();

    // max
    red[tid] = sc[tid];
    __syncthreads();
    for (int off = 128; off > 0; off >>= 1) {
        if (tid < off) red[tid] = fmaxf(red[tid], red[tid + off]);
        __syncthreads();
    }
    float mx = red[0];
    __syncthreads();

    // exp + sum
    float e = (tid < SEQ) ? __expf(sc[tid] - mx) : 0.f;
    red[tid] = e;
    __syncthreads();
    for (int off = 128; off > 0; off >>= 1) {
        if (tid < off) red[tid] += red[tid + off];
        __syncthreads();
    }
    float inv = 1.f / red[0];
    __syncthreads();
    sc[tid] = e;
    __syncthreads();

    // weighted sum of keys, read coalesced straight from HBM:
    // group g handles S range [g*100, g*100+100), lanes sweep d = 0..127.
    const int d = tid & 127;
    const int g = tid >> 7;
    const float* hkb = hk + (size_t)b * SEQ * DIM;
    float psum = 0.f;
    for (int s = g * 100; s < g * 100 + 100; ++s)
        psum += sc[s] * hkb[s * DIM + d];
    red[tid] = psum;
    __syncthreads();
    if (tid < DIM)
        out[(size_t)b * DIM + tid] = (red[tid] + red[tid + 128]) * inv;
}

// ---------------------------------------------------------------------------
extern "C" void kernel_launch(void* const* d_in, const int* in_sizes, int n_in,
                              void* d_out, int out_size, void* d_ws, size_t ws_size,
                              hipStream_t stream)
{
    (void)in_sizes; (void)n_in; (void)out_size; (void)ws_size;

    const float* qg    = (const float*)d_in[0];
    const float* hk    = (const float*)d_in[1];
    const int*   maskp = (const int*)  d_in[2];
    const float* W1    = (const float*)d_in[3];
    const float* b1    = (const float*)d_in[4];
    const float* W2    = (const float*)d_in[5];
    const float* b2    = (const float*)d_in[6];
    const float* W3    = (const float*)d_in[7];
    const float* b3    = (const float*)d_in[8];
    float*       outp  = (float*)d_out;

    // Workspace layout: bf16-swizzled weights, then the raw score matrix.
    unsigned short* w1sw = (unsigned short*)d_ws;            // 512*256 bf16
    unsigned short* w2sw = w1sw + FEAT * H1;                 // 256*128 bf16
    float* scoresbuf = (float*)(w2sw + H1 * H2);             // 2048*200 f32

    // Pre-swizzle weights into WMMA B-fragment layout (bf16).
    int total = FEAT * H1 + H1 * H2;                         // 163840
    prep_swizzle<<<(total + 255) / 256, 256, 0, stream>>>(W1, W2, w1sw, w2sw);

    // Scores: one wave32 per (b, S-tile).
    attn_scores<<<BATCH * NTILE, 32, 0, stream>>>(
        qg, hk, b1, b2, W3, b3, w1sw, w2sw, scoresbuf);

    // Softmax + weighted sum: one block per batch row.
    attn_softmax<<<BATCH, 256, 0, stream>>>(hk, maskp, scoresbuf, outp);
}
